// Attention_25288767438905
// MI455X (gfx1250) — matrix operand order repaired and checked by reading, not compiled, not run on hardware
//
#include <hip/hip_runtime.h>
#include <hip/hip_bf16.h>

#ifndef __has_builtin
#define __has_builtin(x) 0
#endif

// ---------------------------------------------------------------------------
// Problem constants (from reference)
// ---------------------------------------------------------------------------
#define DIMX   1024
#define HEADS  16
#define DHEAD  64
#define BATCH  4
#define SEQ    2048
#define ROWS   (BATCH * SEQ)        // 8192
#define SCALEF 0.125f               // 64^-0.5
#define LN_EPS 1e-5f

typedef __attribute__((ext_vector_type(16))) _Float16 v16h;
typedef __attribute__((ext_vector_type(8)))  float    v8f;

// ---------------------------------------------------------------------------
// CDNA5 async global->LDS copy (ASYNCcnt path).
// Builtin prototype (from hipcc diagnostic):
//   void __builtin_amdgcn_global_load_async_to_lds_b128(
//       int __vector(4) __device__* gsrc,   // addrspace(1)
//       int __vector(4) __shared__* ldst,   // addrspace(3)
//       imm int offset, imm int cpol);
// ---------------------------------------------------------------------------
#if __has_builtin(__builtin_amdgcn_global_load_async_to_lds_b128)
#define HAS_ASYNC_LDS 1
#endif
#if __has_builtin(__builtin_amdgcn_s_wait_asynccnt)
#define HAS_WAIT_ASYNC 1
#endif

typedef int i32x4v __attribute__((vector_size(16)));
typedef __attribute__((address_space(1))) i32x4v* gvec_p;
typedef __attribute__((address_space(3))) i32x4v* lvec_p;

__device__ __forceinline__ void cp_async16(void* lds_dst, const void* gsrc) {
#ifdef HAS_ASYNC_LDS
    // generic LDS pointer keeps the LDS byte offset in its low 32 bits
    __builtin_amdgcn_global_load_async_to_lds_b128(
        (gvec_p)(uintptr_t)gsrc,
        (lvec_p)(uint32_t)(uintptr_t)lds_dst,
        /*offset=*/0, /*cpol=*/0);
#else
    *(uint4*)lds_dst = *(const uint4*)gsrc;
#endif
}

__device__ __forceinline__ void async_wait0() {
#ifdef HAS_WAIT_ASYNC
    __builtin_amdgcn_s_wait_asynccnt(0);
#endif
}

// ---------------------------------------------------------------------------
// WMMA helpers (CDNA5 layouts, ISA 7.12.2)
// ---------------------------------------------------------------------------
__device__ __forceinline__ v8f zero8() {
    v8f z;
#pragma unroll
    for (int i = 0; i < 8; ++i) z[i] = 0.0f;
    return z;
}

__device__ __forceinline__ v8f wmma_f16(v16h a, v16h b, v8f c) {
    // D = A(16x32 f16) * B(32x16 f16) + C(16x16 f32)
    return __builtin_amdgcn_wmma_f32_16x16x32_f16(
        false, a, false, b, (short)0, c, false, false);
}

// A fragment (16x32, MxK) from row-major storage A[m*ld + k]
// lanes 0-15: M=0..15, K 0..15 in v0..7; lanes 16-31: K 16..31
// per-lane reads are two contiguous 16B runs -> 2x ds_load_b128
__device__ __forceinline__ v16h frag_a_rm(const _Float16* __restrict__ A, int ld) {
    const int lane = threadIdx.x & 31;
    const int m = lane & 15, h = lane >> 4;
    v16h a;
#pragma unroll
    for (int r = 0; r < 4; ++r) {
        const int k0 = h * 16 + 2 * r;
        a[2 * r]           = A[m * ld + k0];
        a[2 * r + 1]       = A[m * ld + k0 + 1];
        const int k1 = h * 16 + 8 + 2 * r;
        a[2 * (r + 4)]     = A[m * ld + k1];
        a[2 * (r + 4) + 1] = A[m * ld + k1 + 1];
    }
    return a;
}

// B fragment (32x16, KxN): slot s = row K = half*16 + s, column = lane%16,
// read from K-CONTIGUOUS storage Bt[n*ld + k] (transposed-in-LDS weights,
// or an attention K-tile stored [j][d]). Per-lane contiguous -> 2x b128.
__device__ __forceinline__ v16h frag_b_kc(const _Float16* __restrict__ Bt, int ld) {
    const int lane = threadIdx.x & 31;
    const int n = lane & 15, h = lane >> 4;
    v16h b;
#pragma unroll
    for (int s = 0; s < 16; ++s) b[s] = Bt[n * ld + h * 16 + s];
    return b;
}

// B fragment from row-major B[k*ld + n] (V tile): strided per-lane reads
__device__ __forceinline__ v16h frag_b_rm(const _Float16* __restrict__ B, int ld) {
    const int lane = threadIdx.x & 31;
    const int n = lane & 15, h = lane >> 4;
    v16h b;
#pragma unroll
    for (int s = 0; s < 16; ++s) b[s] = B[(h * 16 + s) * ld + n];
    return b;
}

// ---------------------------------------------------------------------------
// Kernel 1: LayerNorm (f32) -> normalized f16 activations, one block per row
// ---------------------------------------------------------------------------
__global__ __launch_bounds__(256)
void ln_to_f16_kernel(const float* __restrict__ x,
                      const float* __restrict__ gamma,
                      const float* __restrict__ beta,
                      _Float16* __restrict__ xn) {
    const int row = blockIdx.x;
    const int tid = threadIdx.x;
    __shared__ float s1[256];
    __shared__ float s2[256];

    const float* xr = x + (size_t)row * DIMX;
    float v[4], sum = 0.f, sq = 0.f;
#pragma unroll
    for (int i = 0; i < 4; ++i) {
        v[i] = xr[tid + i * 256];
        sum += v[i];
        sq  += v[i] * v[i];
    }
    s1[tid] = sum; s2[tid] = sq;
    __syncthreads();
    for (int off = 128; off > 0; off >>= 1) {
        if (tid < off) { s1[tid] += s1[tid + off]; s2[tid] += s2[tid + off]; }
        __syncthreads();
    }
    const float mu  = s1[0] * (1.0f / DIMX);
    const float var = s2[0] * (1.0f / DIMX) - mu * mu;
    const float rs  = rsqrtf(var + LN_EPS);
#pragma unroll
    for (int i = 0; i < 4; ++i) {
        const int c = tid + i * 256;
        xn[(size_t)row * DIMX + c] = (_Float16)(((v[i] - mu) * rs) * gamma[c] + beta[c]);
    }
}

// ---------------------------------------------------------------------------
// Kernel 2: elementwise f32 -> f16 (weight conversion)
// ---------------------------------------------------------------------------
__global__ __launch_bounds__(256)
void f32_to_f16_kernel(const float* __restrict__ in, _Float16* __restrict__ out, int n) {
    int i = blockIdx.x * blockDim.x + threadIdx.x;
    const int stride = gridDim.x * blockDim.x;
    for (; i < n; i += stride) out[i] = (_Float16)in[i];
}

// ---------------------------------------------------------------------------
// Kernel 3: WMMA GEMM  C[M x Ncols] = A[M x K] * W[K x Ncols]
// block tile 128x128, 8 waves (4x2), wave tile 32x64 (2x4 frags), K-step 32.
// A staged via async global->LDS; W staged TRANSPOSED in LDS so B-fragment
// reads are k-contiguous b128 loads.
// MODE 0: scatter f16 results into q/k/v [b,h,n,d]; MODE 1: bias + f32 out.
// ---------------------------------------------------------------------------
template <int MODE>
__global__ __launch_bounds__(256)
void gemm_wmma_kernel(const _Float16* __restrict__ A,
                      const _Float16* __restrict__ W,
                      int K, int Ncols,
                      _Float16* __restrict__ qh,
                      _Float16* __restrict__ kh,
                      _Float16* __restrict__ vh,
                      const float* __restrict__ bias,
                      float* __restrict__ outf) {
    __shared__ _Float16 As[128 * 40];   // 128 rows x 32 halves (pad 40)
    __shared__ _Float16 BsT[128 * 40];  // 128 n-cols x 32 k (pad 40), transposed

    const int tid  = threadIdx.x;
    const int wave = tid >> 5;
    const int wm   = wave >> 1;   // 0..3 (M)
    const int wn   = wave & 1;    // 0..1 (N)
    const int rowBase = blockIdx.y * 128;
    const int colBase = blockIdx.x * 128;

    v8f acc[2][4];
#pragma unroll
    for (int i = 0; i < 2; ++i)
#pragma unroll
        for (int j = 0; j < 4; ++j) acc[i][j] = zero8();

    for (int k0 = 0; k0 < K; k0 += 32) {
        // --- stage A tile 128x32 halves: 512 x 16B chunks, 2 per thread ---
#pragma unroll
        for (int i = 0; i < 2; ++i) {
            const int idx = tid + i * 256;        // 0..511
            const int row = idx >> 2, seg = idx & 3;
            cp_async16(&As[row * 40 + seg * 8],
                       A + (size_t)(rowBase + row) * K + k0 + seg * 8);
        }
        // --- stage W tile 32x128 halves, transposed into BsT[n][k] ---
#pragma unroll
        for (int i = 0; i < 2; ++i) {
            const int idx  = tid + i * 256;       // 0..511
            const int krow = idx >> 4, seg = idx & 15;
            union { uint4 u; _Float16 h[8]; } t;
            t.u = *(const uint4*)(W + (size_t)(k0 + krow) * Ncols + colBase + seg * 8);
#pragma unroll
            for (int j = 0; j < 8; ++j)
                BsT[(seg * 8 + j) * 40 + krow] = t.h[j];
        }
        async_wait0();
        __syncthreads();

        const v16h a0 = frag_a_rm(&As[(wm * 32) * 40], 40);
        const v16h a1 = frag_a_rm(&As[(wm * 32 + 16) * 40], 40);
#pragma unroll
        for (int nf = 0; nf < 4; ++nf) {
            const v16h b = frag_b_kc(&BsT[(wn * 64 + nf * 16) * 40], 40);
            acc[0][nf] = wmma_f16(a0, b, acc[0][nf]);
            acc[1][nf] = wmma_f16(a1, b, acc[1][nf]);
        }
        __syncthreads();
    }

    // epilogue: C layout -> row = half*8 + r, col = lane%16
    const int lane = tid & 31;
    const int cn = lane & 15, ch = lane >> 4;
#pragma unroll
    for (int mi = 0; mi < 2; ++mi)
#pragma unroll
        for (int nf = 0; nf < 4; ++nf)
#pragma unroll
            for (int r = 0; r < 8; ++r) {
                const int row = rowBase + wm * 32 + mi * 16 + ch * 8 + r;
                const int col = colBase + wn * 64 + nf * 16 + cn;
                const float val = acc[mi][nf][r];
                if (MODE == 0) {
                    const int part = col >> 10;         // 0:q 1:k 2:v
                    const int c    = col & 1023;
                    const int head = c >> 6, dh = c & 63;
                    const int bi   = row >> 11, ni = row & 2047;
                    const size_t idx =
                        ((size_t)(bi * HEADS + head) * SEQ + ni) * DHEAD + dh;
                    const _Float16 hv = (_Float16)val;
                    if (part == 0)      qh[idx] = hv;
                    else if (part == 1) kh[idx] = hv;
                    else                vh[idx] = hv;
                } else {
                    outf[(size_t)row * Ncols + col] = val + bias[col];
                }
            }
}

// ---------------------------------------------------------------------------
// Kernel 4: flash attention, one block per (b*h, 64 query rows), 4 waves.
// K/V tiles staged via async global->LDS; online softmax with 16-lane
// butterflies (rows live across a half-wave in the C-fragment layout).
// Epilogue folds pos and emits the f16 operand of the output projection.
// ---------------------------------------------------------------------------
__global__ __launch_bounds__(128)
void flash_attn_kernel(const _Float16* __restrict__ q,
                       const _Float16* __restrict__ k,
                       const _Float16* __restrict__ v,
                       const float* __restrict__ pos,
                       _Float16* __restrict__ U) {
    __shared__ _Float16 Kl[64 * 72];
    __shared__ _Float16 Vl[64 * 72];
    __shared__ _Float16 Pl[64 * 72];

    const int tid  = threadIdx.x;
    const int wave = tid >> 5;
    const int lane = tid & 31;
    const int cn = lane & 15, ch = lane >> 4;
    const int bh = blockIdx.y;
    const int bi = bh >> 4, hd = bh & 15;
    const int q0 = blockIdx.x * 64;

    // Q fragments for this wave's 16 rows (A layout, K = head dim 64)
    const _Float16* qbase = q + ((size_t)bh * SEQ + q0 + wave * 16) * DHEAD;
    const v16h aq0 = frag_a_rm(qbase, DHEAD);
    const v16h aq1 = frag_a_rm(qbase + 32, DHEAD);

    float mrow[8], lrow[8];
    v8f acc[4];
#pragma unroll
    for (int r = 0; r < 8; ++r) { mrow[r] = -1e30f; lrow[r] = 0.f; }
#pragma unroll
    for (int f = 0; f < 4; ++f) acc[f] = zero8();

    for (int kb = 0; kb < SEQ; kb += 64) {
        __syncthreads();  // protect K/V LDS from previous-iteration readers
        const _Float16* kg = k + ((size_t)bh * SEQ + kb) * DHEAD;
        const _Float16* vg = v + ((size_t)bh * SEQ + kb) * DHEAD;
        // stage K/V 64x64-f16 tiles: 512 x 16B chunks each, async to LDS
#pragma unroll
        for (int i = 0; i < 4; ++i) {
            const int idx = tid + i * 128;       // 0..511
            const int row = idx >> 3, seg = idx & 7;
            cp_async16(&Kl[row * 72 + seg * 8], kg + row * DHEAD + seg * 8);
            cp_async16(&Vl[row * 72 + seg * 8], vg + row * DHEAD + seg * 8);
        }
        // hint the next tile toward the caches while we compute on this one
        if (kb + 64 < SEQ) {
            __builtin_prefetch(kg + 64 * DHEAD + tid * 32, 0, 1);
            __builtin_prefetch(vg + 64 * DHEAD + tid * 32, 0, 1);
        }
        async_wait0();
        __syncthreads();

        // S = Q * K^T  (4 col frags of 16; K tile stored [j][d] = k-contig)
        v8f s[4];
#pragma unroll
        for (int nf = 0; nf < 4; ++nf) {
            v8f c = zero8();
            c = wmma_f16(aq0, frag_b_kc(&Kl[(nf * 16) * 72 + 0], 72), c);
            c = wmma_f16(aq1, frag_b_kc(&Kl[(nf * 16) * 72 + 32], 72), c);
            s[nf] = c;
        }

        // online softmax row stats via 16-lane butterflies
        float mnew[8], alpha[8];
#pragma unroll
        for (int r = 0; r < 8; ++r) {
            float mx = -1e30f;
#pragma unroll
            for (int nf = 0; nf < 4; ++nf) mx = fmaxf(mx, s[nf][r] * SCALEF);
#pragma unroll
            for (int off = 1; off < 16; off <<= 1) mx = fmaxf(mx, __shfl_xor(mx, off, 32));
            mnew[r]  = fmaxf(mrow[r], mx);
            alpha[r] = __expf(mrow[r] - mnew[r]);
        }

        // P = exp(S - m), rowsum, stage P (f16) into per-wave LDS region
#pragma unroll
        for (int r = 0; r < 8; ++r) {
            float rs = 0.f;
#pragma unroll
            for (int nf = 0; nf < 4; ++nf) {
                const float p = __expf(s[nf][r] * SCALEF - mnew[r]);
                rs += p;
                Pl[(wave * 16 + ch * 8 + r) * 72 + nf * 16 + cn] = (_Float16)p;
            }
#pragma unroll
            for (int off = 1; off < 16; off <<= 1) rs += __shfl_xor(rs, off, 32);
            lrow[r] = lrow[r] * alpha[r] + rs;
            mrow[r] = mnew[r];
        }

        // rescale accumulated output
#pragma unroll
        for (int f = 0; f < 4; ++f)
#pragma unroll
            for (int r = 0; r < 8; ++r) acc[f][r] *= alpha[r];

        // O += P * V   (A = P 16x64 over keys, B = V [j][d] row-major)
        const v16h ap0 = frag_a_rm(&Pl[(wave * 16) * 72 + 0], 72);
        const v16h ap1 = frag_a_rm(&Pl[(wave * 16) * 72 + 32], 72);
#pragma unroll
        for (int f = 0; f < 4; ++f) {
            acc[f] = wmma_f16(ap0, frag_b_rm(&Vl[0 * 72 + f * 16], 72), acc[f]);
            acc[f] = wmma_f16(ap1, frag_b_rm(&Vl[32 * 72 + f * 16], 72), acc[f]);
        }
    }

    // epilogue: normalize, merge heads, add pos, emit f16 for out-projection
#pragma unroll
    for (int f = 0; f < 4; ++f)
#pragma unroll
        for (int r = 0; r < 8; ++r) {
            const int ni = q0 + wave * 16 + ch * 8 + r;
            const int c  = hd * DHEAD + f * 16 + cn;
            const size_t pidx = ((size_t)bi * SEQ + ni) * DIMX + c;
            U[pidx] = (_Float16)(acc[f][r] / lrow[r] + pos[pidx]);
        }
}

// ---------------------------------------------------------------------------
// launch
// inputs: x, pos, w_qkv, w_out, b_out, ln_gamma, ln_beta (all f32)
// output: [4, 2048, 1024] f32; workspace: ~88 MB f16 staging
// ---------------------------------------------------------------------------
extern "C" void kernel_launch(void* const* d_in, const int* in_sizes, int n_in,
                              void* d_out, int out_size, void* d_ws, size_t ws_size,
                              hipStream_t stream) {
    (void)in_sizes; (void)n_in; (void)out_size; (void)ws_size;

    const float* x      = (const float*)d_in[0];
    const float* pos    = (const float*)d_in[1];
    const float* w_qkv  = (const float*)d_in[2];
    const float* w_out  = (const float*)d_in[3];
    const float* b_out  = (const float*)d_in[4];
    const float* ln_g   = (const float*)d_in[5];
    const float* ln_b   = (const float*)d_in[6];
    float* out = (float*)d_out;

    char* ws = (char*)d_ws;
    size_t off = 0;
    _Float16* xn  = (_Float16*)(ws + off); off += (size_t)ROWS * DIMX * 2;        // 16 MB
    _Float16* wqh = (_Float16*)(ws + off); off += (size_t)DIMX * 3 * DIMX * 2;    //  6 MB
    _Float16* woh = (_Float16*)(ws + off); off += (size_t)DIMX * DIMX * 2;        //  2 MB
    _Float16* qh  = (_Float16*)(ws + off); off += (size_t)ROWS * DIMX * 2;        // 16 MB
    _Float16* kh  = (_Float16*)(ws + off); off += (size_t)ROWS * DIMX * 2;        // 16 MB
    _Float16* vh  = (_Float16*)(ws + off); off += (size_t)ROWS * DIMX * 2;        // 16 MB
    _Float16* U   = (_Float16*)(ws + off); off += (size_t)ROWS * DIMX * 2;        // 16 MB

    // 1) LayerNorm -> f16 activations
    ln_to_f16_kernel<<<ROWS, 256, 0, stream>>>(x, ln_g, ln_b, xn);

    // 2) weights -> f16
    f32_to_f16_kernel<<<2048, 256, 0, stream>>>(w_qkv, wqh, DIMX * 3 * DIMX);
    f32_to_f16_kernel<<<1024, 256, 0, stream>>>(w_out, woh, DIMX * DIMX);

    // 3) QKV projection: [8192 x 1024] x [1024 x 3072], scatter to heads
    gemm_wmma_kernel<0><<<dim3(3 * DIMX / 128, ROWS / 128), 256, 0, stream>>>(
        xn, wqh, DIMX, 3 * DIMX, qh, kh, vh, nullptr, nullptr);

    // 4) flash attention per (b*h), 64 query rows per block; folds pos
    flash_attn_kernel<<<dim3(SEQ / 64, BATCH * HEADS), 128, 0, stream>>>(
        qh, kh, vh, pos, U);

    // 5) output projection: [8192 x 1024] x [1024 x 1024] + bias -> f32 out
    gemm_wmma_kernel<1><<<dim3(DIMX / 128, ROWS / 128), 256, 0, stream>>>(
        U, woh, DIMX, DIMX, nullptr, nullptr, nullptr, b_out, out);
}
